// SparseConv2D_11776800325854
// MI455X (gfx1250) — compile-verified, matrix operand order checked by reading, not run on hardware
//
#include <hip/hip_runtime.h>

// ---------------------------------------------------------------------------
// SparseConv2D (dense weights here) as im2col GEMM on CDNA5 WMMA.
// fp32 inputs -> bf16 hi/lo split (3-term) -> v_wmma_f32_16x16x32_bf16 -> fp32.
// ---------------------------------------------------------------------------

typedef __bf16        v16bf __attribute__((ext_vector_type(16)));
typedef float         v8f   __attribute__((ext_vector_type(8)));
typedef unsigned int  u32x4 __attribute__((ext_vector_type(4)));

#define NBATCH 16
#define CIN    64
#define H_IN   112
#define W_IN   112
#define COUT   128
#define HO     110
#define WO     110
#define KK     576            // CIN * 3 * 3
#define KSTEPS 18             // KK / 32
#define MTILES 8              // COUT / 16
#define A_ELEMS (MTILES * KSTEPS * 32 * 16)   // bf16 elements per (hi|lo) array

// B-panel in LDS: [kstep(18)][subtile(4)][hi/lo(2)] fragments of 1024 B each.
// Fragment layout: two 512 B half-blocks; lane stride 16 B inside each block
// (16 B aligned ds_load_b128, optimal 2-pass banking).
#define FRAG_BYTES 1024
#define LDS_BYTES  (KSTEPS * 4 * 2 * FRAG_BYTES)   // 147456 B -> 2 WGs / WGP

// ---------------------------------------------------------------------------
// Prep: split fp32 weights into bf16 hi/lo, pre-swizzled to the 16-bit
// A-matrix 16x32 WMMA fragment layout (ISA 7.12.2): lane half h holds
// K = {0..7}+8h (VGPR0-3) and {16..23}+8h (VGPR4-7), 2 elems per VGPR.
// ---------------------------------------------------------------------------
__global__ void pack_weights_kernel(const float* __restrict__ w,
                                    __bf16* __restrict__ ahi,
                                    __bf16* __restrict__ alo) {
  int idx = blockIdx.x * 256 + threadIdx.x;
  if (idx >= A_ELEMS) return;
  int j    = idx & 15;                 // element within lane's 32B chunk
  int lane = (idx >> 4) & 31;
  int s    = (idx >> 9) % KSTEPS;      // k-step
  int mt   = idx / (16 * 32 * KSTEPS); // Cout tile
  int half = lane >> 4;
  int v    = j >> 1;                   // VGPR index 0..7
  int hw   = j & 1;                    // halfword
  int kr   = ((v >= 4) ? 16 : 0) + ((v & 3) << 1) + hw + 8 * half;
  int m    = mt * 16 + (lane & 15);
  int k    = s * 32 + kr;
  float f  = w[m * KK + k];
  __bf16 hi = (__bf16)f;
  __bf16 lo = (__bf16)(f - (float)hi);
  ahi[idx] = hi;
  alo[idx] = lo;
}

// ---------------------------------------------------------------------------
// Main kernel: one workgroup = 8 waves = (n, ho, 64-wide wo chunk).
// Wave w computes Cout rows [16w, 16w+16) x 64 output pixels.
// ---------------------------------------------------------------------------
__global__ void __launch_bounds__(256)
conv_gemm_kernel(const float* __restrict__ x,
                 const __bf16* __restrict__ ahi,
                 const __bf16* __restrict__ alo,
                 const float* __restrict__ bias,
                 float* __restrict__ out) {
  __shared__ char smem[LDS_BYTES];

  const int tid  = threadIdx.x;
  const int lane = tid & 31;
  const int wave = tid >> 5;            // 0..7 -> Cout tile
  const int wo0  = blockIdx.x * 64;
  const int ho   = blockIdx.y;
  const int n    = blockIdx.z;

  // ---- Build im2col B-panel (bf16 hi/lo, fragment-swizzled) in LDS. ----
  // Valid conv: ho+kh <= 111 < 112 and wo+kw <= 111 < 112 -> no pad checks.
  {
    const int colAll = tid & 63;                 // pixel within chunk
    const int wo     = wo0 + colAll;
    const int woc    = (wo < WO) ? wo : (WO - 1); // clamp edge (garbage cols)
    const int t      = colAll >> 4;              // subtile
    const int col    = colAll & 15;              // N within subtile
    const int kBase  = tid >> 6;                 // 0..3
    const float* xrow = x + ((size_t)n * CIN) * (H_IN * W_IN);
    for (int i = 0; i < KK / 4; ++i) {
      int k   = kBase + (i << 2);                // patch index (C,kh,kw)
      int c   = k / 9;
      int rem = k - 9 * c;
      int kh  = rem / 3;
      int kw  = rem - 3 * kh;
      float f = xrow[((size_t)c * H_IN + (ho + kh)) * W_IN + (woc + kw)];
      __bf16 hi = (__bf16)f;
      __bf16 lo = (__bf16)(f - (float)hi);
      int s    = k >> 5;
      int kr   = k & 31;
      int half = ((kr & 15) >= 8) ? 1 : 0;       // lane half
      int blk  = (kr >= 16) ? 1 : 0;             // 512B half-block
      int v2   = (kr & 7) >> 1;                  // dword within half-block
      int hw   = kr & 1;
      int base = ((s * 4 + t) * 2) * FRAG_BYTES + blk * 512 +
                 (col + 16 * half) * 16 + v2 * 4 + hw * 2;
      *(__bf16*)(smem + base)              = hi;   // hi fragment
      *(__bf16*)(smem + base + FRAG_BYTES) = lo;   // lo fragment
    }
  }
  __syncthreads();

  // ---- WMMA main loop: D += Ah*Bh + Al*Bh + Ah*Bl (fp32-accurate split) ----
  union BF { v16bf v; u32x4 q[2]; };
  const v16bf* Ah = (const v16bf*)ahi + (size_t)(wave * KSTEPS) * 32 + lane;
  const v16bf* Al = (const v16bf*)alo + (size_t)(wave * KSTEPS) * 32 + lane;

  v8f acc[4] = {};
  for (int s = 0; s < KSTEPS; ++s) {
    v16bf a_hi = Ah[s * 32];
    v16bf a_lo = Al[s * 32];
    for (int t = 0; t < 4; ++t) {
      BF bh, bl;
      const char* p = smem + ((s * 4 + t) * 2) * FRAG_BYTES + lane * 16;
      bh.q[0] = *(const u32x4*)(p);
      bh.q[1] = *(const u32x4*)(p + 512);
      bl.q[0] = *(const u32x4*)(p + FRAG_BYTES);
      bl.q[1] = *(const u32x4*)(p + FRAG_BYTES + 512);
      acc[t] = __builtin_amdgcn_wmma_f32_16x16x32_bf16(
          false, a_hi, false, bh.v, (short)0, acc[t], false, false);
      acc[t] = __builtin_amdgcn_wmma_f32_16x16x32_bf16(
          false, a_lo, false, bh.v, (short)0, acc[t], false, false);
      acc[t] = __builtin_amdgcn_wmma_f32_16x16x32_bf16(
          false, a_hi, false, bl.v, (short)0, acc[t], false, false);
    }
  }

  // ---- Epilogue: bias + store (C/D layout: VGPR r -> M = r + 8*(lane/16),
  //      N = lane%16). Skip garbage columns past WO. ----
  const int half = lane >> 4;
  const int m0   = wave * 16;
  for (int t = 0; t < 4; ++t) {
    int wo = wo0 + t * 16 + (lane & 15);
    if (wo < WO) {
      for (int r = 0; r < 8; ++r) {
        int cout = m0 + r + 8 * half;
        out[(((size_t)n * COUT + cout) * HO + ho) * WO + wo] =
            acc[t][r] + bias[cout];
      }
    }
  }
}

// ---------------------------------------------------------------------------
extern "C" void kernel_launch(void* const* d_in, const int* in_sizes, int n_in,
                              void* d_out, int out_size, void* d_ws, size_t ws_size,
                              hipStream_t stream) {
  const float* x    = (const float*)d_in[0];   // [16,64,112,112]
  const float* w    = (const float*)d_in[1];   // [128,576]
  const float* bias = (const float*)d_in[2];   // [128,1]
  float* out        = (float*)d_out;           // [16,128,110,110]

  __bf16* ahi = (__bf16*)d_ws;                 // A_ELEMS bf16
  __bf16* alo = ahi + A_ELEMS;                 // A_ELEMS bf16 (total ~288 KB)

  pack_weights_kernel<<<(A_ELEMS + 255) / 256, 256, 0, stream>>>(w, ahi, alo);

  dim3 grid((WO + 63) / 64, HO, NBATCH);       // (2, 110, 16)
  conv_gemm_kernel<<<grid, 256, 0, stream>>>(x, ahi, alo, bias, out);
}
